// LPCNetModelBunch_39427799777470
// MI455X (gfx1250) — compile-verified, matrix operand order checked by reading, not run on hardware
//
#include <hip/hip_runtime.h>

// ---------------------------------------------------------------------------
// LPCNet forward on gfx1250 (wave32, WMMA). All GEMMs use
// v_wmma_f32_16x16x32_f16 (f16 inputs, f32 accum). Sequential GRU scans keep
// hidden state in LDS; batch dimension provides the WMMA M-tile.
// This revision keeps per-iteration register pressure low (f16 hidden copy in
// LDS, double-buffered GEMM fragments) so the allocator can actually overlap
// VMEM/LDS loads with the WMMA pipe instead of s_wait_loadcnt 0x0 per WMMA.
// ---------------------------------------------------------------------------

typedef _Float16 v16h __attribute__((ext_vector_type(16)));
typedef _Float16 v8h  __attribute__((ext_vector_type(8)));
typedef float    v8f  __attribute__((ext_vector_type(8)));

#define WMMA16(A, B, C) \
  __builtin_amdgcn_wmma_f32_16x16x32_f16(false, (A), false, (B), (short)0, (C), false, false)

// Model constants
#define BATCH 64
#define TFULL 1760
#define TR    880           // T / R
#define MROWS (BATCH * TR)  // 56320 GEMM rows
#define KA    896           // GRU-A input width
#define NA    1152          // 3*H1
#define H1    384
#define KB    512           // GRU-B input width (H1 + DF)
#define NB    48            // 3*H2
#define H2    16
#define KC    160           // padded mdense K (144 -> 160)
#define NC    256           // ULAW
#define DF    128
#define NFRM  11
#define HSTR  392           // padded f16 h row stride (halves): banks spread

__device__ __forceinline__ float sigmoidf_(float x) { return 1.0f / (1.0f + __expf(-x)); }

// Load one 16x32 WMMA fragment row-slice for this lane (global or LDS).
// p points at (row-of-this-lane)*ld + k0 + (lane>=16 ? 8 : 0). Must be 16B aligned.
template <typename P>
__device__ __forceinline__ v16h load_frag16(const P* p) {
  v8h lo = *(const v8h*)(p);
  v8h hh = *(const v8h*)(p + 16);
  v16h f;
#pragma unroll
  for (int q = 0; q < 8; ++q) { f[q] = lo[q]; f[q + 8] = hh[q]; }
  return f;
}

// ---------------------------------------------------------------------------
// 1) Frame feature path: embed pitch, conv1(84->128,k3), conv2(128->128,k3),
//    fd1, fd2 (all tanh). One block per (batch, out-frame). Tiny: VALU fp32.
// ---------------------------------------------------------------------------
__global__ void frame_kernel(const float* __restrict__ features,
                             const int* __restrict__ periods,
                             const float* __restrict__ embed_pitch,
                             const float* __restrict__ c1w, const float* __restrict__ c1b,
                             const float* __restrict__ c2w, const float* __restrict__ c2b,
                             const float* __restrict__ fd1w, const float* __restrict__ fd1b,
                             const float* __restrict__ fd2w, const float* __restrict__ fd2b,
                             float* __restrict__ fout) {
  int b = blockIdx.x / NFRM;
  int t = blockIdx.x % NFRM;
  __shared__ float cat[5 * 84];
  __shared__ float c1s[3 * 128];
  __shared__ float x2s[128];
  __shared__ float x3s[128];
  int tid = threadIdx.x;  // 128

  for (int e = tid; e < 5 * 84; e += 128) {
    int tt = e / 84, ii = e % 84;
    int ft = t + tt;  // 0..14
    cat[e] = (ii < 20) ? features[(b * 15 + ft) * 20 + ii]
                       : embed_pitch[periods[b * 15 + ft] * 64 + (ii - 20)];
  }
  __syncthreads();
  int o = tid;
  for (int kk = 0; kk < 3; ++kk) {
    float s = c1b[o];
    for (int k = 0; k < 3; ++k)
      for (int i = 0; i < 84; ++i)
        s += cat[(kk + k) * 84 + i] * c1w[o * 252 + i * 3 + k];
    c1s[kk * 128 + o] = tanhf(s);
  }
  __syncthreads();
  {
    float s = c2b[o];
    for (int k = 0; k < 3; ++k)
      for (int i = 0; i < 128; ++i)
        s += c1s[k * 128 + i] * c2w[o * 384 + i * 3 + k];
    x2s[o] = tanhf(s);
  }
  __syncthreads();
  {
    float s = fd1b[o];
    for (int i = 0; i < 128; ++i) s += x2s[i] * fd1w[o * 128 + i];
    x3s[o] = tanhf(s);
  }
  __syncthreads();
  {
    float s = fd2b[o];
    for (int i = 0; i < 128; ++i) s += x3s[i] * fd2w[o * 128 + i];
    fout[(b * NFRM + t) * 128 + o] = tanhf(s);
  }
}

// ---------------------------------------------------------------------------
// 2) Weight f32 -> f16 pack with optional K zero-padding. dst is [N][Kp].
// ---------------------------------------------------------------------------
__global__ void pack_w_kernel(const float* __restrict__ src, _Float16* __restrict__ dst,
                              int Ks, int Kp) {
  long n = blockIdx.x;
  for (int k = threadIdx.x; k < Kp; k += blockDim.x)
    dst[n * Kp + k] = (k < Ks) ? (_Float16)src[n * Ks + k] : (_Float16)0.0f;
}

// ---------------------------------------------------------------------------
// 3) Pack GRU-A input rows (f16): [sig-embed x6 segs of 128 | frame feat 128]
// ---------------------------------------------------------------------------
__global__ void pack_a1_kernel(const int* __restrict__ in_data,
                               const float* __restrict__ embed_sig,
                               const float* __restrict__ f,
                               _Float16* __restrict__ A1) {
  long m = blockIdx.x;
  int b = (int)(m / TR), t = (int)(m % TR);
  int e = threadIdx.x;  // 128
  _Float16* row = A1 + m * KA;
#pragma unroll
  for (int seg = 0; seg < 6; ++seg) {
    int s = seg / 3, c = seg % 3;
    int idx = in_data[((long)b * TFULL + 2 * t + s) * 3 + c];
    row[seg * 128 + e] = (_Float16)embed_sig[idx * 128 + e];
  }
  row[768 + e] = (_Float16)f[((long)b * NFRM + t / 80) * 128 + e];
}

// Fill GRU-B input cols [384,512) with the repeated frame features (f16).
__global__ void pack_a2_rep_kernel(const float* __restrict__ f, _Float16* __restrict__ A2) {
  long m = blockIdx.x;
  int b = (int)(m / TR), t = (int)(m % TR);
  int e = threadIdx.x;  // 128
  A2[m * KB + H1 + e] = (_Float16)f[((long)b * NFRM + t / 80) * 128 + e];
}

// Fill mdense ctx cols [16,144) with sig-embed of pred_exc, zero-pad [144,160).
__global__ void pack_ctx_kernel(const int* __restrict__ targets,
                                const float* __restrict__ embed_sig,
                                _Float16* __restrict__ ctx) {
  long m = blockIdx.x;
  int b = (int)(m / TR), t = (int)(m % TR);
  int e = threadIdx.x;  // 128
  int idx = targets[(long)b * TFULL + 2 * t];
  ctx[m * KC + 16 + e] = (_Float16)embed_sig[idx * 128 + e];
  if (e < 16) ctx[m * KC + 144 + e] = (_Float16)0.0f;
}

// ---------------------------------------------------------------------------
// 4) Generic WMMA GEMM: C[M][N] = A[M][K] * W[N][K]^T. One wave per block,
//    NT 16-col tiles per wave (A fragment reused NT times). The k-loop is
//    double-buffered (fragments for k0+32 issued before the WMMAs for k0).
//    NT chosen so live VGPRs (acc 8*NT + 2*(8 + 8*NT)) stay well under 256.
// ---------------------------------------------------------------------------
template <int NT>
__global__ void gemm_wmma_kernel(const _Float16* __restrict__ A,
                                 const _Float16* __restrict__ W,
                                 float* __restrict__ C, int N, int K) {
  int m0 = blockIdx.x * 16;
  int n0 = blockIdx.y * 16 * NT;
  int lane = threadIdx.x;
  int li = lane & 15, hb = lane >> 4;
  v8f acc[NT];
#pragma unroll
  for (int i = 0; i < NT; ++i) acc[i] = (v8f){};
  const _Float16* arow = A + (long)(m0 + li) * K + hb * 8;
  const _Float16* wrow = W + (long)(n0 + li) * K + hb * 8;  // nt offsets are +16*K

  // Preload k0 = 0 fragments.
  v16h af = load_frag16(arow);
  v16h bf[NT];
#pragma unroll
  for (int nt = 0; nt < NT; ++nt) bf[nt] = load_frag16(wrow + (long)nt * 16 * K);

  for (int k0 = 0; k0 < K; k0 += 32) {
    // Issue next iteration's loads (re-load current on the final iteration —
    // always a valid address, avoids a branch).
    int k1 = (k0 + 32 < K) ? (k0 + 32) : k0;
    v16h afn = load_frag16(arow + k1);
    v16h bfn[NT];
#pragma unroll
    for (int nt = 0; nt < NT; ++nt) bfn[nt] = load_frag16(wrow + (long)nt * 16 * K + k1);
    // Consume current fragments.
#pragma unroll
    for (int nt = 0; nt < NT; ++nt) acc[nt] = WMMA16(af, bf[nt], acc[nt]);
    af = afn;
#pragma unroll
    for (int nt = 0; nt < NT; ++nt) bf[nt] = bfn[nt];
  }
#pragma unroll
  for (int nt = 0; nt < NT; ++nt)
#pragma unroll
    for (int r = 0; r < 8; ++r)
      C[(long)(m0 + r + hb * 8) * N + n0 + nt * 16 + li] = acc[nt][r];
}

// ---------------------------------------------------------------------------
// 5) GRU-A sequential scan. 4 blocks x 16 batch rows, 8 waves. Per step:
//    gh = h @ Whh^T via WMMA (each wave: 9 N-tiles x 12 K-tiles), then gate
//    math on 256 threads. Hidden state lives in LDS twice: f32 master for the
//    gate update and a padded f16 copy that A fragments are rebuilt from via
//    two ds_load_b128 per fragment — per-iteration register pressure is ~24
//    VGPRs, so the unrolled k-chain can keep several B loads in flight.
//    Writes g1 as f16 into A2[:, 0:384].
// ---------------------------------------------------------------------------
__global__ void gru_a_scan_kernel(const float* __restrict__ xWa,
                                  const _Float16* __restrict__ Whh,   // [1152][384] f16
                                  const float* __restrict__ bih,
                                  const float* __restrict__ bhh,
                                  _Float16* __restrict__ A2) {
  extern __shared__ float smem[];
  float*    hbuf = smem;                         // 16*384 f32
  float*    gh   = smem + 16 * H1;               // 16*1152 f32
  _Float16* h16  = (_Float16*)(gh + 16 * NA);    // 16*HSTR f16 (padded rows)
  int b0 = blockIdx.x * 16;
  int tid = threadIdx.x;         // 256
  int wave = tid >> 5, lane = tid & 31;
  int li = lane & 15, hb = lane >> 4;

  for (int e = tid; e < 16 * H1; e += 256) hbuf[e] = 0.0f;
  for (int e = tid; e < 16 * HSTR; e += 256) h16[e] = (_Float16)0.0f;
  __syncthreads();

  for (int t = 0; t < TR; ++t) {
    const _Float16* hrow = h16 + li * HSTR + hb * 8;
    // Each wave: 9 output tiles of 16 columns.
    for (int tt = 0; tt < 9; ++tt) {
      int n0 = (wave * 9 + tt) * 16;
      const _Float16* wrow = Whh + (long)(n0 + li) * H1 + hb * 8;
      v8f acc = (v8f){};
#pragma unroll
      for (int kt = 0; kt < 12; ++kt) {
        v16h a = load_frag16(hrow + kt * 32);   // ds_load_b128 x2
        v16h b = load_frag16(wrow + kt * 32);   // global_load_b128 x2
        acc = WMMA16(a, b, acc);
      }
#pragma unroll
      for (int r = 0; r < 8; ++r)
        gh[(r + hb * 8) * NA + n0 + li] = acc[r];
    }
    __syncthreads();
    // Gate update: 16*384 hidden elements across 256 threads.
    for (int e = tid; e < 16 * H1; e += 256) {
      int i = e / H1, j = e % H1;
      long m = (long)(b0 + i) * TR + t;
      const float* xw = xWa + m * NA;
      float ghr = gh[i * NA + j] + bhh[j];
      float ghz = gh[i * NA + H1 + j] + bhh[H1 + j];
      float ghn = gh[i * NA + 2 * H1 + j] + bhh[2 * H1 + j];
      float r = sigmoidf_(xw[j] + bih[j] + ghr);
      float z = sigmoidf_(xw[H1 + j] + bih[H1 + j] + ghz);
      float n = tanhf(xw[2 * H1 + j] + bih[2 * H1 + j] + r * ghn);
      float hn = (1.0f - z) * n + z * hbuf[e];
      hbuf[e] = hn;
      _Float16 hf = (_Float16)hn;
      h16[i * HSTR + j] = hf;
      A2[m * KB + j] = hf;
    }
    __syncthreads();
  }
}

// ---------------------------------------------------------------------------
// 6) GRU-B sequential scan (H2=16, tiny): one block, 256 threads, pure VALU.
//    Writes g2 as f16 into ctx[:, 0:16].
// ---------------------------------------------------------------------------
__global__ void gru_b_scan_kernel(const float* __restrict__ xWb,
                                  const float* __restrict__ Whh,  // [48][16] f32
                                  const float* __restrict__ bih,
                                  const float* __restrict__ bhh,
                                  _Float16* __restrict__ ctx) {
  __shared__ float h[BATCH * H2];
  __shared__ float gh[BATCH * NB];
  __shared__ float w[NB * H2];
  int tid = threadIdx.x;  // 256
  for (int e = tid; e < NB * H2; e += 256) w[e] = Whh[e];
  for (int e = tid; e < BATCH * H2; e += 256) h[e] = 0.0f;
  __syncthreads();
  for (int t = 0; t < TR; ++t) {
    for (int e = tid; e < BATCH * NB; e += 256) {
      int b = e / NB, n = e % NB;
      float s = bhh[n];
#pragma unroll
      for (int k = 0; k < H2; ++k) s += h[b * H2 + k] * w[n * H2 + k];
      gh[e] = s;
    }
    __syncthreads();
    for (int e = tid; e < BATCH * H2; e += 256) {
      int b = e / H2, j = e % H2;
      long m = (long)b * TR + t;
      const float* xw = xWb + m * NB;
      float r = sigmoidf_(xw[j] + bih[j] + gh[b * NB + j]);
      float z = sigmoidf_(xw[H2 + j] + bih[H2 + j] + gh[b * NB + H2 + j]);
      float n = tanhf(xw[2 * H2 + j] + bih[2 * H2 + j] + r * gh[b * NB + 2 * H2 + j]);
      float hn = (1.0f - z) * n + z * h[e];
      h[e] = hn;
      ctx[m * KC + j] = (_Float16)hn;
    }
    __syncthreads();
  }
}

// ---------------------------------------------------------------------------
// 7) Dual-mdense output: 4 WMMA GEMMs (md1_w1, md1_w2, md2_w1, md2_w2) over
//    one padded ctx (K=160), fused tanh/scale, interleaved (o1,o2) store.
//    k-loop double-buffered (1 A + 4 B fragments in flight).
// ---------------------------------------------------------------------------
__global__ void mdense_out_kernel(const _Float16* __restrict__ ctx,
                                  const _Float16* __restrict__ Wmd,  // [4][256][160]
                                  const float* __restrict__ b1a, const float* __restrict__ b2a,
                                  const float* __restrict__ f1a, const float* __restrict__ f2a,
                                  const float* __restrict__ b1b, const float* __restrict__ b2b,
                                  const float* __restrict__ f1b, const float* __restrict__ f2b,
                                  float* __restrict__ out) {
  int m0 = blockIdx.x * 16;
  int c0 = blockIdx.y * 16;
  int lane = threadIdx.x;
  int li = lane & 15, hb = lane >> 4;
  v8f acc[4];
#pragma unroll
  for (int i = 0; i < 4; ++i) acc[i] = (v8f){};

  const _Float16* arow = ctx + (long)(m0 + li) * KC + hb * 8;
  const _Float16* wrow = Wmd + (long)(c0 + li) * KC + hb * 8;  // mt offsets +NC*KC

  v16h af = load_frag16(arow);
  v16h bf[4];
#pragma unroll
  for (int mt = 0; mt < 4; ++mt) bf[mt] = load_frag16(wrow + (long)mt * NC * KC);

  for (int k0 = 0; k0 < KC; k0 += 32) {
    int k1 = (k0 + 32 < KC) ? (k0 + 32) : k0;
    v16h afn = load_frag16(arow + k1);
    v16h bfn[4];
#pragma unroll
    for (int mt = 0; mt < 4; ++mt) bfn[mt] = load_frag16(wrow + (long)mt * NC * KC + k1);
#pragma unroll
    for (int mt = 0; mt < 4; ++mt) acc[mt] = WMMA16(af, bf[mt], acc[mt]);
    af = afn;
#pragma unroll
    for (int mt = 0; mt < 4; ++mt) bf[mt] = bfn[mt];
  }

  int col = c0 + li;
#pragma unroll
  for (int r = 0; r < 8; ++r) {
    int row = m0 + r + hb * 8;
    int b = row / TR, t = row % TR;
    float o1 = tanhf(acc[0][r] + b1a[col]) * f1a[col] + tanhf(acc[1][r] + b2a[col]) * f2a[col];
    float o2 = tanhf(acc[2][r] + b1b[col]) * f1b[col] + tanhf(acc[3][r] + b2b[col]) * f2b[col];
    out[((long)b * TFULL + 2 * t) * NC + col] = o1;
    out[((long)b * TFULL + 2 * t + 1) * NC + col] = o2;
  }
}

// ---------------------------------------------------------------------------
extern "C" void kernel_launch(void* const* d_in, const int* in_sizes, int n_in,
                              void* d_out, int out_size, void* d_ws, size_t ws_size,
                              hipStream_t stream) {
  const int*   in_data     = (const int*)d_in[0];
  const float* features    = (const float*)d_in[1];
  const int*   periods     = (const int*)d_in[2];
  const int*   targets     = (const int*)d_in[3];
  const float* embed_pitch = (const float*)d_in[4];
  const float* embed_sig   = (const float*)d_in[5];
  const float* conv1_w = (const float*)d_in[6];
  const float* conv1_b = (const float*)d_in[7];
  const float* conv2_w = (const float*)d_in[8];
  const float* conv2_b = (const float*)d_in[9];
  const float* fd1_w = (const float*)d_in[10];
  const float* fd1_b = (const float*)d_in[11];
  const float* fd2_w = (const float*)d_in[12];
  const float* fd2_b = (const float*)d_in[13];
  const float* ga_wih = (const float*)d_in[14];
  const float* ga_whh = (const float*)d_in[15];
  const float* ga_bih = (const float*)d_in[16];
  const float* ga_bhh = (const float*)d_in[17];
  const float* gb_wih = (const float*)d_in[18];
  const float* gb_whh = (const float*)d_in[19];
  const float* gb_bih = (const float*)d_in[20];
  const float* gb_bhh = (const float*)d_in[21];
  const float* md1_w1 = (const float*)d_in[22];
  const float* md1_w2 = (const float*)d_in[23];
  const float* md1_b1 = (const float*)d_in[24];
  const float* md1_b2 = (const float*)d_in[25];
  const float* md1_f1 = (const float*)d_in[26];
  const float* md1_f2 = (const float*)d_in[27];
  const float* md2_w1 = (const float*)d_in[28];
  const float* md2_w2 = (const float*)d_in[29];
  const float* md2_b1 = (const float*)d_in[30];
  const float* md2_b2 = (const float*)d_in[31];
  const float* md2_f1 = (const float*)d_in[32];
  const float* md2_f2 = (const float*)d_in[33];
  float* out = (float*)d_out;

  // Workspace carve-out (256B aligned segments).
  char* ws = (char*)d_ws;
  size_t off = 0;
  auto carve = [&](size_t bytes) -> char* {
    char* p = ws + off;
    off = (off + bytes + 255) & ~(size_t)255;
    return p;
  };
  float*    f_f32  = (float*)carve((size_t)BATCH * NFRM * DF * 4);
  _Float16* A1     = (_Float16*)carve((size_t)MROWS * KA * 2);
  _Float16* WaIH   = (_Float16*)carve((size_t)NA * KA * 2);
  _Float16* WaHH   = (_Float16*)carve((size_t)NA * H1 * 2);
  float*    xWa    = (float*)carve((size_t)MROWS * NA * 4);
  _Float16* A2     = (_Float16*)carve((size_t)MROWS * KB * 2);
  _Float16* WbIH   = (_Float16*)carve((size_t)NB * KB * 2);
  float*    xWb    = (float*)carve((size_t)MROWS * NB * 4);
  _Float16* ctx    = (_Float16*)carve((size_t)MROWS * KC * 2);
  _Float16* Wmd    = (_Float16*)carve((size_t)4 * NC * KC * 2);

  // 1) frame features
  frame_kernel<<<BATCH * NFRM, 128, 0, stream>>>(features, periods, embed_pitch,
      conv1_w, conv1_b, conv2_w, conv2_b, fd1_w, fd1_b, fd2_w, fd2_b, f_f32);

  // 2) weight packs (f32 -> f16, pad K where needed)
  pack_w_kernel<<<NA, 128, 0, stream>>>(ga_wih, WaIH, KA, KA);
  pack_w_kernel<<<NA, 128, 0, stream>>>(ga_whh, WaHH, H1, H1);
  pack_w_kernel<<<NB, 128, 0, stream>>>(gb_wih, WbIH, KB, KB);
  pack_w_kernel<<<NC, 128, 0, stream>>>(md1_w1, Wmd + 0 * NC * KC, H2, KC);
  pack_w_kernel<<<NC, 128, 0, stream>>>(md1_w2, Wmd + 1 * NC * KC, H2, KC);
  pack_w_kernel<<<NC, 128, 0, stream>>>(md2_w1, Wmd + 2 * NC * KC, H2 + DF, KC);
  pack_w_kernel<<<NC, 128, 0, stream>>>(md2_w2, Wmd + 3 * NC * KC, H2 + DF, KC);

  // 3) GRU-A input pack + big GEMM: xWa = A1 @ ga_wih^T (NT=6 -> 96 cols/blk)
  pack_a1_kernel<<<MROWS, 128, 0, stream>>>(in_data, embed_sig, f_f32, A1);
  {
    dim3 g(MROWS / 16, NA / 96);
    gemm_wmma_kernel<6><<<g, 32, 0, stream>>>(A1, WaIH, xWa, NA, KA);
  }

  // 4) GRU-A sequential scan (4 blocks x 16 batch rows); writes A2[:,0:384]
  {
    size_t shmem = (size_t)(16 * H1 + 16 * NA) * sizeof(float)
                 + (size_t)(16 * HSTR) * sizeof(_Float16);  // 110848 B
    gru_a_scan_kernel<<<4, 256, shmem, stream>>>(xWa, WaHH, ga_bih, ga_bhh, A2);
  }

  // 5) GRU-B input GEMM: xWb = A2 @ gb_wih^T
  pack_a2_rep_kernel<<<MROWS, 128, 0, stream>>>(f_f32, A2);
  {
    dim3 g(MROWS / 16, 1);  // NB = 48 = 3 tiles
    gemm_wmma_kernel<3><<<g, 32, 0, stream>>>(A2, WbIH, xWb, NB, KB);
  }

  // 6) ctx pack + GRU-B scan (writes ctx[:,0:16])
  pack_ctx_kernel<<<MROWS, 128, 0, stream>>>(targets, embed_sig, ctx);
  gru_b_scan_kernel<<<1, 256, 0, stream>>>(xWb, gb_whh, gb_bih, gb_bhh, ctx);

  // 7) dual mdense + interleaved output
  {
    dim3 g(MROWS / 16, NC / 16);
    mdense_out_kernel<<<g, 32, 0, stream>>>(ctx, Wmd,
        md1_b1, md1_b2, md1_f1, md1_f2,
        md2_b1, md2_b2, md2_f1, md2_f2, out);
  }
}
